// Encoder_82068235092496
// MI455X (gfx1250) — matrix operand order repaired and checked by reading, not compiled, not run on hardware
//
#include <hip/hip_runtime.h>
#include <cmath>

// ---------------------------------------------------------------------------
// CDNA5 (gfx1250) wave32 WMMA types
// ---------------------------------------------------------------------------
typedef _Float16 f16;
typedef __attribute__((ext_vector_type(16))) _Float16 v16h;
typedef __attribute__((ext_vector_type(8)))  _Float16 v8h;
typedef __attribute__((ext_vector_type(8)))  float    v8f;
typedef __attribute__((ext_vector_type(4)))  unsigned u32x4;
typedef __attribute__((ext_vector_type(8)))  int      i32x8;
typedef __attribute__((ext_vector_type(4)))  int      i32x4;

union V16U { v16h v; v8h h[2]; };

__device__ __forceinline__ v8f wmma_f16(v16h a, v16h b, v8f c) {
  // D = A(16x32 f16) x B(32x16 f16) + C(16x16 f32)
  return __builtin_amdgcn_wmma_f32_16x16x32_f16(
      /*neg_a=*/false, a, /*neg_b=*/false, b,
      /*c_mod=*/(short)0, c, /*reuse_a=*/false, /*reuse_b=*/false);
}

// LDS byte offset of a generic pointer to __shared__ data: generic LDS
// addresses carry the wave-relative offset in addr[31:0].
__device__ __forceinline__ unsigned lds_lo(const void* p) {
  return (unsigned)(unsigned long long)p;
}

// gfx1250 async global->LDS copy (ASYNCcnt) : 16 bytes per lane
__device__ __forceinline__ void async_ld_b128(unsigned lds, const void* gptr) {
  unsigned long long ga = (unsigned long long)gptr;
  asm volatile("global_load_async_to_lds_b128 %0, %1, off"
               :: "v"(lds), "v"(ga) : "memory");
}
__device__ __forceinline__ void wait_async0() {
  asm volatile("s_wait_asynccnt 0" ::: "memory");
}

// ===========================================================================
// GEMM: out[M,N] = act(A[M,K] * W[K,N] + bias[N]) (+ res[M,N])
// f32 in HBM, f16 tiles in LDS (double-buffered), f32 WMMA accumulate.
// Optional f16 output (used for Q/K/V so attention can DMA raw f16).
// Requires M%128==0, N%128==0, K%32==0 (true for all call sites here).
// ===========================================================================
#define BM 128
#define BN 128
#define BK 32

__global__ __launch_bounds__(256) void gemm_kernel(
    const float* __restrict__ A, const float* __restrict__ W,
    const float* __restrict__ bias, const float* res,
    float* out, f16* out16, int M, int N, int K, int act)
{
  __shared__ __attribute__((aligned(16))) f16 As[2][BM][BK];   // [m][k]
  __shared__ __attribute__((aligned(16))) f16 Bs[2][BN][BK];   // transposed [n][k]

  const int tid  = threadIdx.x;
  const int n0   = blockIdx.x * BN;
  const int m0   = blockIdx.y * BM;
  const int wave = tid >> 5;
  const int lane = tid & 31;
  const int wm   = wave & 3;        // 4 wave-rows of 32
  const int wn   = wave >> 2;       // 2 wave-cols of 64
  const int lm   = lane & 15;
  const int lh   = lane >> 4;
  const int KT   = K / BK;

  // per-thread load slots (float4 x4 for A, x4 for B)
  int aM[4], aK[4], bK[4], bN[4];
#pragma unroll
  for (int j = 0; j < 4; ++j) {
    int base = tid * 4 + 1024 * j;      // 4096 elements per tile
    aM[j] = base >> 5;  aK[j] = base & 31;
    bK[j] = base >> 7;  bN[j] = base & 127;
  }

  float4 areg[4], breg[4];
  auto issue_loads = [&](int k0) {
#pragma unroll
    for (int j = 0; j < 4; ++j) {
      areg[j] = *(const float4*)(&A[(size_t)(m0 + aM[j]) * K + k0 + aK[j]]);
      breg[j] = *(const float4*)(&W[(size_t)(k0 + bK[j]) * N + n0 + bN[j]]);
    }
  };
  auto store_tile = [&](int buf) {
#pragma unroll
    for (int j = 0; j < 4; ++j) {
      As[buf][aM[j]][aK[j] + 0] = (f16)areg[j].x;
      As[buf][aM[j]][aK[j] + 1] = (f16)areg[j].y;
      As[buf][aM[j]][aK[j] + 2] = (f16)areg[j].z;
      As[buf][aM[j]][aK[j] + 3] = (f16)areg[j].w;
      Bs[buf][bN[j] + 0][bK[j]] = (f16)breg[j].x;
      Bs[buf][bN[j] + 1][bK[j]] = (f16)breg[j].y;
      Bs[buf][bN[j] + 2][bK[j]] = (f16)breg[j].z;
      Bs[buf][bN[j] + 3][bK[j]] = (f16)breg[j].w;
    }
  };

  v8f acc[2][4] = {};

  issue_loads(0);
  store_tile(0);
  __syncthreads();

  for (int kt = 0; kt < KT; ++kt) {
    const int cur = kt & 1;
    if (kt + 1 < KT) issue_loads((kt + 1) * BK);   // overlap with WMMA below
    if (kt + 2 < KT) {                             // L2 prefetch 2 tiles ahead
      __builtin_prefetch(&A[(size_t)(m0 + (tid >> 1)) * K + (kt + 2) * BK], 0, 3);
      __builtin_prefetch(&W[(size_t)((kt + 2) * BK + (tid >> 3)) * N + n0 + (tid & 7) * 16], 0, 3);
    }

    v16h af[2], bf[4];
#pragma unroll
    for (int im = 0; im < 2; ++im) {
      const f16* pa = &As[cur][wm * 32 + im * 16 + lm][0];
      V16U u;
      u.h[0] = *(const v8h*)(pa + lh * 16);       // K = lh*16 + 0..7
      u.h[1] = *(const v8h*)(pa + lh * 16 + 8);   // K = lh*16 + 8..15
      af[im] = u.v;
    }
#pragma unroll
    for (int jn = 0; jn < 4; ++jn) {
      const f16* pb = &Bs[cur][wn * 64 + jn * 16 + lm][lh * 16]; // K = lh*16+0..15
      V16U u;
      u.h[0] = *(const v8h*)pb;
      u.h[1] = *(const v8h*)(pb + 8);
      bf[jn] = u.v;
    }
#pragma unroll
    for (int im = 0; im < 2; ++im)
#pragma unroll
      for (int jn = 0; jn < 4; ++jn)
        acc[im][jn] = wmma_f16(af[im], bf[jn], acc[im][jn]);

    if (kt + 1 < KT) store_tile(cur ^ 1);
    __syncthreads();
  }

  // ---- epilogue: bias, gelu, residual; f32 or f16 store ----
#pragma unroll
  for (int im = 0; im < 2; ++im) {
#pragma unroll
    for (int jn = 0; jn < 4; ++jn) {
      int gm = m0 + wm * 32 + im * 16 + lh * 8;
      int gn = n0 + wn * 64 + jn * 16 + lm;
      float bv = bias[gn];
#pragma unroll
      for (int r = 0; r < 8; ++r) {
        float v = acc[im][jn][r] + bv;
        if (act) v = 0.5f * v * (1.0f + erff(v * 0.70710678118654752f)); // exact gelu
        size_t idx = (size_t)(gm + r) * N + gn;
        if (res) v += res[idx];
        if (out16) out16[idx] = (f16)v;
        else       out[idx]   = v;
      }
    }
  }
}

// ===========================================================================
// LayerNorm over last dim (768) ; one block per row
// ===========================================================================
__global__ __launch_bounds__(256) void ln_kernel(
    const float* __restrict__ x, const float* __restrict__ g,
    const float* __restrict__ b, float* __restrict__ out)
{
  __shared__ float red[256];
  const int t = threadIdx.x;
  const float* xr = x + (size_t)blockIdx.x * 768;
  float v0 = xr[t], v1 = xr[t + 256], v2 = xr[t + 512];

  red[t] = v0 + v1 + v2;
  __syncthreads();
  for (int off = 128; off > 0; off >>= 1) {
    if (t < off) red[t] += red[t + off];
    __syncthreads();
  }
  float mean = red[0] * (1.0f / 768.0f);
  __syncthreads();

  float d0 = v0 - mean, d1 = v1 - mean, d2 = v2 - mean;
  red[t] = d0 * d0 + d1 * d1 + d2 * d2;
  __syncthreads();
  for (int off = 128; off > 0; off >>= 1) {
    if (t < off) red[t] += red[t + off];
    __syncthreads();
  }
  float rstd = rsqrtf(red[0] * (1.0f / 768.0f) + 1e-5f);

  float* orow = out + (size_t)blockIdx.x * 768;
  orow[t]       = g[t]       * d0 * rstd + b[t];
  orow[t + 256] = g[t + 256] * d1 * rstd + b[t + 256];
  orow[t + 512] = g[t + 512] * d2 * rstd + b[t + 512];
}

// ===========================================================================
// Attention: per (b,h): softmax(Q Kt / 8) V over f16 Q/K/V [B*S, 768].
// Block = 128 thr = 4 waves; Q tile via TDM tensor_load_to_lds; K/V chunks
// via per-lane async global->LDS b128; all matmuls on WMMA, f32 softmax.
// ===========================================================================
#define NB 16
#define NH 12

__global__ __launch_bounds__(128) void attn_kernel(
    const f16* __restrict__ Q, const f16* __restrict__ K,
    const f16* __restrict__ V, float* __restrict__ O, int Sq, int Sk)
{
  __shared__ __attribute__((aligned(16))) f16 Qs[64][64];
  __shared__ __attribute__((aligned(16))) f16 Ks[32][64];
  __shared__ __attribute__((aligned(16))) f16 Vs[32][64];
  __shared__ __attribute__((aligned(16))) f16 Ps[4][16][32];

  const int tid = threadIdx.x, lane = tid & 31, wave = tid >> 5;
  const int lm = lane & 15, lh = lane >> 4;
  const int tiles = Sq >> 6;
  const int bh = blockIdx.x / tiles, tile = blockIdx.x % tiles;
  const int b = bh / NH, h = bh % NH;

  // ---- Q tile (64 rows x 64 f16, row stride 768) via Tensor Data Mover ----
  const size_t qbase = (size_t)(b * Sq + tile * 64) * 768 + h * 64;
  if (tid < 32) {   // one wave issues the TDM descriptor (EXEC-independent op)
    unsigned long long ga = (unsigned long long)(const void*)(Q + qbase);
    u32x4 g0;
    g0[0] = 1u;                                    // count=1, load, user mode
    g0[1] = lds_lo(&Qs[0][0]);                     // lds_addr (bytes)
    g0[2] = (unsigned)(ga & 0xffffffffu);          // global_addr[31:0]
    g0[3] = (unsigned)((ga >> 32) & 0x01ffffffu)   // global_addr[56:32]
          | (2u << 30);                            // type=2 ("image")
    i32x8 g1;
    g1[0] = (1 << 16);                             // data_size=1 -> 2 bytes
    g1[1] = 0;                                     // td0[15:0]=0 (td0=1<<20)
    g1[2] = 0x10;                                  // td0[31:16] | td1[15:0]=0
    g1[3] = (64 << 16) | 0x10;                     // tile_dim0=64 | td1[31:16]
    g1[4] = 64;                                    // tile_dim1=64, tile_dim2=0
    g1[5] = 768;                                   // tensor_dim0_stride=768
    g1[6] = 0;
    g1[7] = 0;
    i32x4 gz4 = {0, 0, 0, 0};
    i32x8 gz8 = {0, 0, 0, 0, 0, 0, 0, 0};
    __builtin_amdgcn_tensor_load_to_lds(g0, g1, gz4, gz4, gz8, 0);
    __builtin_amdgcn_s_wait_tensorcnt(0);
  }
  __syncthreads();

  v16h aq[2];
#pragma unroll
  for (int c = 0; c < 2; ++c) {
    const f16* p = &Qs[wave * 16 + lm][c * 32];
    V16U u;
    u.h[0] = *(const v8h*)(p + lh * 16);
    u.h[1] = *(const v8h*)(p + lh * 16 + 8);
    aq[c] = u.v;
  }

  v8f o[4] = {};
  float mi[8], li[8];
#pragma unroll
  for (int r = 0; r < 8; ++r) { mi[r] = -1e30f; li[r] = 0.0f; }

  const size_t kbase = (size_t)(b * Sk) * 768 + h * 64;
  const int nchunk = Sk >> 5;
  for (int kc = 0; kc < nchunk; ++kc) {
    __syncthreads();
    // ---- 32x64 f16 K and V chunks via async global->LDS (16B per lane) ----
#pragma unroll
    for (int j = 0; j < 2; ++j) {
      int idx = tid + 128 * j;          // 256 x 16B segments per matrix
      int r = idx >> 3, seg = (idx & 7) * 8;
      size_t src = kbase + (size_t)(kc * 32 + r) * 768 + seg;
      async_ld_b128(lds_lo(&Ks[r][seg]), K + src);
      async_ld_b128(lds_lo(&Vs[r][seg]), V + src);
    }
    wait_async0();
    __syncthreads();

    // ---- S(16x32) = Q(16x64) . Kt(64x32) : 4 WMMAs ----
    v8f s[2];
#pragma unroll
    for (int sub = 0; sub < 2; ++sub) {
      s[sub] = (v8f){};
#pragma unroll
      for (int c = 0; c < 2; ++c) {
        const f16* p = &Ks[sub * 16 + lm][c * 32 + lh * 16];
        V16U u;
        u.h[0] = *(const v8h*)p;
        u.h[1] = *(const v8h*)(p + 8);
        s[sub] = wmma_f16(aq[c], u.v, s[sub]);
      }
    }

    // ---- online softmax (rows r+8*lh; cols across 16-lane halves) ----
#pragma unroll
    for (int r = 0; r < 8; ++r) {
      float a0 = s[0][r] * 0.125f;      // 1/sqrt(64)
      float a1 = s[1][r] * 0.125f;
      float mx = fmaxf(a0, a1);
      for (int off = 1; off < 16; off <<= 1) mx = fmaxf(mx, __shfl_xor(mx, off, 32));
      float mn = fmaxf(mi[r], mx);
      float alpha = expf(mi[r] - mn);
      float p0 = expf(a0 - mn);
      float p1 = expf(a1 - mn);
      float rs = p0 + p1;
      for (int off = 1; off < 16; off <<= 1) rs += __shfl_xor(rs, off, 32);
      li[r] = li[r] * alpha + rs;
      mi[r] = mn;
#pragma unroll
      for (int gdim = 0; gdim < 4; ++gdim) o[gdim][r] *= alpha;
      int prow = r + 8 * lh;
      Ps[wave][prow][lm]      = (f16)p0;
      Ps[wave][prow][16 + lm] = (f16)p1;
    }
    __syncthreads();

    // ---- O(16x64) += P(16x32) . V(32x64) : 4 WMMAs ----
    v16h ap;
    {
      const f16* p = &Ps[wave][lm][0];
      V16U u;
      u.h[0] = *(const v8h*)(p + lh * 16);
      u.h[1] = *(const v8h*)(p + lh * 16 + 8);
      ap = u.v;
    }
#pragma unroll
    for (int gdim = 0; gdim < 4; ++gdim) {
      v16h bv;
      int n = gdim * 16 + lm;
#pragma unroll
      for (int e = 0; e < 16; ++e) bv[e] = Vs[lh * 16 + e][n];
      o[gdim] = wmma_f16(ap, bv, o[gdim]);
    }
  }

  // ---- normalize and write (f32 context for O-projection GEMM) ----
  const size_t obase = (size_t)(b * Sq + tile * 64 + wave * 16) * 768 + h * 64;
#pragma unroll
  for (int gdim = 0; gdim < 4; ++gdim)
#pragma unroll
    for (int r = 0; r < 8; ++r) {
      int row = 8 * lh + r;
      O[obase + (size_t)row * 768 + gdim * 16 + lm] = o[gdim][r] / li[r];
    }
}

// ===========================================================================
// Host orchestration
// ===========================================================================
namespace {

struct PLin { const float *W, *b; };
struct PLN  { const float *g, *b; };
struct PMHA { PLin q, k, v, o; };
struct PFF  { PLin fc1, fc2; };
struct PDual { PLN ln1, ln2, ln3, ln4; PMHA a1, a2, a3; PFF f1, f2; };
struct PLayer { PLN ln1, ln2, ln3; PMHA a1, a2; PDual dual; PFF ff; };

struct Scratch { f16 *qh, *kh, *vh; float *ctx, *h; };

inline void gemm(hipStream_t st, const float* A, const float* W, const float* bias,
                 const float* res, float* out, f16* out16,
                 int M, int N, int K, int act) {
  dim3 grid(N / BN, M / BM);
  gemm_kernel<<<grid, 256, 0, st>>>(A, W, bias, res, out, out16, M, N, K, act);
}

inline void layernorm(hipStream_t st, const float* x, const PLN& p, float* out, int rows) {
  ln_kernel<<<rows, 256, 0, st>>>(x, p.g, p.b, out);
}

inline void mha_run(hipStream_t st, const PMHA& p, const float* q_in, int Sq,
                    const float* kv_in, int Sk, const float* res, float* out,
                    const Scratch& ws) {
  const int Mq = NB * Sq, Mk = NB * Sk;
  gemm(st, q_in,  p.q.W, p.q.b, nullptr, nullptr, ws.qh, Mq, 768, 768, 0);
  gemm(st, kv_in, p.k.W, p.k.b, nullptr, nullptr, ws.kh, Mk, 768, 768, 0);
  gemm(st, kv_in, p.v.W, p.v.b, nullptr, nullptr, ws.vh, Mk, 768, 768, 0);
  attn_kernel<<<NB * NH * (Sq / 64), 128, 0, st>>>(ws.qh, ws.kh, ws.vh, ws.ctx, Sq, Sk);
  gemm(st, ws.ctx, p.o.W, p.o.b, res, out, nullptr, Mq, 768, 768, 0);
}

inline void ff_run(hipStream_t st, const PFF& p, const float* x2, int M,
                   const float* res, float* out, const Scratch& ws) {
  gemm(st, x2, p.fc1.W, p.fc1.b, nullptr, ws.h, nullptr, M, 3072, 768, 1);  // gelu
  gemm(st, ws.h, p.fc2.W, p.fc2.b, res, out, nullptr, M, 768, 3072, 0);
}

} // namespace

extern "C" void kernel_launch(void* const* d_in, const int* in_sizes, int n_in,
                              void* d_out, int out_size, void* d_ws, size_t ws_size,
                              hipStream_t stream) {
  (void)in_sizes; (void)n_in; (void)out_size; (void)ws_size;

  // ---- parse params (dict-insertion order: v_feat, w_feat, s_feat, layers) ----
  int cur = 3;
  auto nxt  = [&]() -> const float* { return (const float*)d_in[cur++]; };
  auto gLin = [&]() { PLin l; l.W = nxt(); l.b = nxt(); return l; };
  auto gLN  = [&]() { PLN l; l.g = nxt(); l.b = nxt(); return l; };
  auto gMHA = [&]() { PMHA m; m.q = gLin(); m.k = gLin(); m.v = gLin(); m.o = gLin(); return m; };
  auto gFF  = [&]() { PFF f; f.fc1 = gLin(); f.fc2 = gLin(); return f; };
  auto gDual = [&]() {
    PDual d;
    d.ln1 = gLN(); d.ln2 = gLN(); d.ln3 = gLN(); d.ln4 = gLN();
    d.a1 = gMHA(); d.a2 = gMHA(); d.a3 = gMHA();
    d.f1 = gFF();  d.f2 = gFF();
    return d;
  };
  auto gLayer = [&]() {
    PLayer L;
    L.ln1 = gLN(); L.ln2 = gLN(); L.ln3 = gLN();
    L.a1 = gMHA(); L.a2 = gMHA();
    L.dual = gDual(); L.ff = gFF();
    return L;
  };
  PLayer layers[4];
  for (int i = 0; i < 4; ++i) layers[i] = gLayer();

  // ---- workspace arena (~245 MB) ----
  const size_t MV = (size_t)NB * 512 * 768;   // 6,291,456
  const size_t MW = (size_t)NB * 128 * 768;   // 1,572,864
  const size_t MH = (size_t)NB * 512 * 3072;  // 25,165,824
  float* p = (float*)d_ws;
  auto arena = [&](size_t nfloats) { float* r = p; p += nfloats; return r; };
  float* A_v   = arena(MV);   // v-stream state
  float* A_vn  = arena(MV);   // v-stream layernorm out
  float* A_e   = arena(MW);   // dual stream x / e_out
  float* A_en  = arena(MW);   // dual layernorm out
  float* A_w   = arena(MW);   // w-stream state
  float* A_wn  = arena(MW);   // ln1(w_feat)
  float* A_s   = arena(MW);   // s-stream state
  Scratch ws;
  ws.qh  = (f16*)arena(MV / 2);
  ws.kh  = (f16*)arena(MV / 2);
  ws.vh  = (f16*)arena(MV / 2);
  ws.ctx = arena(MV);
  ws.h   = arena(MH);

  const float* curV = (const float*)d_in[0];
  const float* curW = (const float*)d_in[1];
  const float* curS = (const float*)d_in[2];

  for (int li = 0; li < 4; ++li) {
    const PLayer& L = layers[li];
    // x2 = ln1(v); x = v + selfattn(x2)
    layernorm(stream, curV, L.ln1, A_vn, NB * 512);
    mha_run(stream, L.a1, A_vn, 512, A_vn, 512, curV, A_v, ws);
    // x2 = ln2(x)  (A_vn persists through dual: dual's v_feat & layer-attn2 q)
    layernorm(stream, A_v, L.ln2, A_vn, NB * 512);

    // ---- dual(x2, w_feat, s_feat) ----
    const PDual& D = L.dual;
    layernorm(stream, curW, D.ln1, A_wn, NB * 128);               // w = ln1(w_feat)
    mha_run(stream, D.a1, A_wn, 128, A_wn, 128, A_wn, A_e, ws);   // x = w + selfattn(w)
    layernorm(stream, A_e, D.ln2, A_en, NB * 128);
    mha_run(stream, D.a2, A_en, 128, A_vn, 512, A_e, A_e, ws);    // cross with v (Sk=512)
    layernorm(stream, A_e, D.ln3, A_en, NB * 128);
    ff_run(stream, D.f1, A_en, NB * 128, A_e, A_w, ws);           // x1 -> next w_feat
    mha_run(stream, D.a3, A_en, 128, curS, 128, A_e, A_e, ws);    // cross with s
    layernorm(stream, A_e, D.ln4, A_en, NB * 128);
    ff_run(stream, D.f2, A_en, NB * 128, A_e, A_s, ws);           // x3 -> next s_feat
    // e_out = A_e

    // x = x + crossattn(x2, e_out)
    mha_run(stream, L.a2, A_vn, 512, A_e, 128, A_v, A_v, ws);
    // out = x + ff(ln3(x))
    layernorm(stream, A_v, L.ln3, A_vn, NB * 512);
    ff_run(stream, L.ff, A_vn, NB * 512, A_v, A_v, ws);

    curV = A_v; curW = A_w; curS = A_s;
  }

  // ---- pack outputs (v_feat, w_feat, s_feat) ----
  float* out = (float*)d_out;
  (void)hipMemcpyAsync(out,           A_v, MV * sizeof(float), hipMemcpyDeviceToDevice, stream);
  (void)hipMemcpyAsync(out + MV,      A_w, MW * sizeof(float), hipMemcpyDeviceToDevice, stream);
  (void)hipMemcpyAsync(out + MV + MW, A_s, MW * sizeof(float), hipMemcpyDeviceToDevice, stream);
}
